// LocalDiffusionMACE_2370821947749
// MI455X (gfx1250) — matrix operand order repaired and checked by reading, not compiled, not run on hardware
//
#include <hip/hip_runtime.h>
#include <hip/hip_bf16.h>
#include <math.h>

// ---------------------------------------------------------------------------
// LocalDiffusionMACE forward for MI455X (gfx1250, wave32, WMMA).
// - Node GEMMs: v_wmma_f32_16x16x32_f16, LDS staged in fragment order so each
//   A/B fragment is a single aligned v16h (2x ds_load_b128) load.
// - Compile-time epilogue flags (bias / accumulate / silu-on-input / sq+cube),
//   with a uniform-branch full-tile fast path (straight-line clause stores).
// - Edge kernel: fused geometry + radial MLP (register-held ew2 columns,
//   v_readlane broadcasts) + f32 atomic scatter of the 4 live SH channels.
// ---------------------------------------------------------------------------

typedef __attribute__((ext_vector_type(16))) _Float16 v16h;
typedef __attribute__((ext_vector_type(8)))  float    v8f;

#define GF_BIAS   1
#define GF_ACC    2
#define GF_SILUX  4
#define GF_SQCUBE 8

__device__ __forceinline__ float silu_f(float x) {
    return x / (1.0f + __expf(-x));
}

__device__ __forceinline__ float bcast_lane(float v, int l) {
    return __builtin_bit_cast(float,
        __builtin_amdgcn_readlane(__builtin_bit_cast(int, v), l));
}

// ---------------------------------------------------------------------------
// Y[M x 64] (+)= op(X[M x 64]) @ W[64 x 64] (+ bias); optional S=Y^2, C=Y^3.
// 256 threads = 8 waves; wave computes a 16x64 C tile: 4 N-tiles x 2 K-steps.
// ---------------------------------------------------------------------------
template<int FLAGS>
__global__ void __launch_bounds__(256)
gemm64_wmma(const float* __restrict__ X, const float* __restrict__ W,
            const float* __restrict__ bias, float* __restrict__ Y,
            float* __restrict__ Sq, float* __restrict__ Cb, int M)
{
    // [waveTile(8) | kstep(2) | lane(32)] x 16 halves  = 16 KB
    __shared__ __align__(32) _Float16 sXf[8 * 2 * 32 * 16];
    // [kstep(2) | ntile(4) | lane(32)] x 16 halves     =  8 KB
    __shared__ __align__(32) _Float16 sWf[2 * 4 * 32 * 16];

    const int tid = threadIdx.x;
    const int m0  = blockIdx.x * 128;

    // ---- W -> LDS in B-fragment order: one 256-thread pass, one v16h store.
    {
        const int slot = tid;             // ks*128 + t*32 + lane
        const int ks   = slot >> 7;
        const int t    = (slot >> 5) & 3;
        const int l    = slot & 31;
        const int col  = t * 16 + (l & 15);
        const int kb16 = (l >> 4) * 8;    // B layout: K = ks*32 + kb16 + (j&7) + (j>=8 ? 16 : 0)
        v16h wf;
#pragma unroll
        for (int j = 0; j < 16; ++j)
            wf[j] = (_Float16)W[(ks * 32 + kb16 + (j & 7) + ((j >> 3) << 4)) * 64 + col];
        ((v16h*)sWf)[slot] = wf;
    }

    // ---- X -> LDS scattered into A-fragment order (coalesced global reads).
    // A layout per lane-half: K 0-7/16-23 (half 0), 8-15/24-31 (half 1).
    for (int idx = tid; idx < 128 * 64; idx += 256) {
        const int r = idx >> 6, c = idx & 63;
        const int gm = m0 + r;
        float v = (gm < M) ? X[(size_t)gm * 64 + c] : 0.0f;
        if constexpr (FLAGS & GF_SILUX) v = silu_f(v);
        const int wvt = r >> 4, l16r = r & 15;
        const int ks = c >> 5, kk = c & 31;
        const int halfk = (kk >> 3) & 1;
        const int j = (kk & 7) + ((kk >= 16) ? 8 : 0);
        const int lanei = halfk * 16 + l16r;
        sXf[(((wvt * 2 + ks) * 32) + lanei) * 16 + j] = (_Float16)v;
    }
    __syncthreads();

    const int wv   = tid >> 5;
    const int lane = tid & 31;
    const v16h* Xf = (const v16h*)sXf;
    const v16h* Wf = (const v16h*)sWf;

    const v16h a0 = Xf[(wv * 2 + 0) * 32 + lane];
    const v16h a1 = Xf[(wv * 2 + 1) * 32 + lane];

    v8f acc[4];
#pragma unroll
    for (int t = 0; t < 4; ++t) acc[t] = (v8f){};

#pragma unroll
    for (int t = 0; t < 4; ++t) {
        const v16h b0 = Wf[(0 * 4 + t) * 32 + lane];
        const v16h b1 = Wf[(1 * 4 + t) * 32 + lane];
        acc[t] = __builtin_amdgcn_wmma_f32_16x16x32_f16(
                     false, a0, false, b0, (short)0, acc[t], false, false);
        acc[t] = __builtin_amdgcn_wmma_f32_16x16x32_f16(
                     false, a1, false, b1, (short)0, acc[t], false, false);
    }

    // ---- Epilogue. C/D layout: VGPR r -> row half*8+r, col = t*16+l16.
    const int half  = lane >> 4;
    const int l16   = lane & 15;
    const int rbase = m0 + wv * 16 + half * 8;

    if (m0 + 128 <= M) {           // uniform full-tile path: no guards
#pragma unroll
        for (int t = 0; t < 4; ++t) {
            const int col = t * 16 + l16;
            float bv = 0.0f;
            if constexpr (FLAGS & GF_BIAS) bv = bias[col];
#pragma unroll
            for (int r = 0; r < 8; ++r) {
                const size_t off = (size_t)(rbase + r) * 64 + col;
                float v = acc[t][r];
                if constexpr (FLAGS & GF_BIAS) v += bv;
                if constexpr (FLAGS & GF_ACC)  v += Y[off];
                Y[off] = v;
                if constexpr (FLAGS & GF_SQCUBE) {
                    const float v2 = v * v;
                    Sq[off] = v2;
                    Cb[off] = v2 * v;
                }
            }
        }
    } else {                       // single partial tile: guarded
#pragma unroll
        for (int t = 0; t < 4; ++t) {
            const int col = t * 16 + l16;
            float bv = 0.0f;
            if constexpr (FLAGS & GF_BIAS) bv = bias[col];
#pragma unroll
            for (int r = 0; r < 8; ++r) {
                if (rbase + r < M) {
                    const size_t off = (size_t)(rbase + r) * 64 + col;
                    float v = acc[t][r];
                    if constexpr (FLAGS & GF_BIAS) v += bv;
                    if constexpr (FLAGS & GF_ACC)  v += Y[off];
                    Y[off] = v;
                    if constexpr (FLAGS & GF_SQCUBE) {
                        const float v2 = v * v;
                        Sq[off] = v2;
                        Cb[off] = v2 * v;
                    }
                }
            }
        }
    }
}

// ---------------------------------------------------------------------------
// Fused edge kernel. One wave per edge (grid-stride); lane owns channels
// h and h+32. ew2 columns live in registers; g broadcast via v_readlane.
// Only SH channels 0..3 are live; 1/AVG_NEIGH folded at the source.
// ---------------------------------------------------------------------------
__global__ void __launch_bounds__(256)
edge_scatter(const int* __restrict__ ei, const float* __restrict__ pos,
             const float* __restrict__ shifts, const float* __restrict__ tnode,
             const float* __restrict__ ew1, const float* __restrict__ eb1,
             const float* __restrict__ ew2, const float* __restrict__ eb2,
             const float* __restrict__ up, float* __restrict__ A,
             int E, int N)
{
    const int lane = threadIdx.x & 31;
    const int wv   = threadIdx.x >> 5;
    const int h0 = lane, h1 = lane + 32;
    const int N64 = N * 64;
    const float s3 = 1.7320508075688772f;

    // Register-resident ew2 columns for this lane's two channels (128 VGPRs).
    float c0[64], c1[64];
#pragma unroll
    for (int hp = 0; hp < 64; ++hp) {
        c0[hp] = ew2[hp * 64 + h0];
        c1[hp] = ew2[hp * 64 + h1];
    }

    const float w1a0 = ew1[h0], w1b0 = ew1[64 + h0], b10 = eb1[h0];
    const float w1a1 = ew1[h1], w1b1 = ew1[64 + h1], b11 = eb1[h1];
    const float b20 = eb2[h0], b21 = eb2[h1];

    for (int e = blockIdx.x * 8 + wv; e < E; e += gridDim.x * 8) {
        const int s = ei[e];
        const int r = ei[E + e];
        float vx = pos[r * 3 + 0] - pos[s * 3 + 0] + shifts[e * 3 + 0];
        float vy = pos[r * 3 + 1] - pos[s * 3 + 1] + shifts[e * 3 + 1];
        float vz = pos[r * 3 + 2] - pos[s * 3 + 2] + shifts[e * 3 + 2];
        float len = sqrtf(vx * vx + vy * vy + vz * vz + 1e-12f);
        float inv = 1.0f / len;
        float x = vx * inv, y = vy * inv, z = vz * inv;
        float tn = tnode[s];

        float g0 = silu_f(len * w1a0 + tn * w1b0 + b10);
        float g1 = silu_f(len * w1a1 + tn * w1b1 + b11);

        float w0 = b20, w1 = b21;
#pragma unroll
        for (int hp = 0; hp < 32; ++hp) {
            const float ga = bcast_lane(g0, hp);
            const float gb = bcast_lane(g1, hp);
            w0 += ga * c0[hp] + gb * c0[hp + 32];
            w1 += ga * c1[hp] + gb * c1[hp + 32];
        }

        const float m0 = up[(size_t)s * 64 + h0] * w0 * 0.0625f;  // 1/AVG_NEIGH
        const float m1 = up[(size_t)s * 64 + h1] * w1 * 0.0625f;

        float* base = A + (size_t)r * 64;
        unsafeAtomicAdd(base + h0,           m0);
        unsafeAtomicAdd(base + h1,           m1);
        unsafeAtomicAdd(base + N64 + h0,     m0 * (s3 * x));
        unsafeAtomicAdd(base + N64 + h1,     m1 * (s3 * x));
        unsafeAtomicAdd(base + 2 * N64 + h0, m0 * (s3 * y));
        unsafeAtomicAdd(base + 2 * N64 + h1, m1 * (s3 * y));
        unsafeAtomicAdd(base + 3 * N64 + h0, m0 * (s3 * z));
        unsafeAtomicAdd(base + 3 * N64 + h1, m1 * (s3 * z));
    }
}

// ---------------------------------------------------------------------------
// Small kernels
// ---------------------------------------------------------------------------
__global__ void fill_zero(float* __restrict__ p, int n)
{
    int i = blockIdx.x * blockDim.x + threadIdx.x;
    if (i < n) p[i] = 0.0f;
}

__global__ void init_nodes(const float* __restrict__ positions,
                           const float* __restrict__ attrs,
                           const float* __restrict__ t, const int* __restrict__ batch,
                           const float* __restrict__ emb_w, const float* __restrict__ emb_b,
                           float* __restrict__ h, float* __restrict__ pos,
                           float* __restrict__ tnode, int N)
{
    int idx = blockIdx.x * blockDim.x + threadIdx.x;
    if (idx >= N * 64) return;
    int n = idx >> 6, k = idx & 63;
    float tn = t[batch[n]];
    float acc = emb_b[k];
#pragma unroll
    for (int j = 0; j < 5; ++j)
        acc += (attrs[n * 5 + j] * 0.25f) * emb_w[j * 64 + k];   // /NORM_FACTOR
    acc += tn * emb_w[5 * 64 + k];
    h[idx] = acc;
    if (k == 0) tnode[n] = tn;
    if (k < 3) pos[n * 3 + k] = positions[n * 3 + k];
}

__global__ void compute_pv(const float* __restrict__ W1, const float* __restrict__ rvec,
                           float* __restrict__ pv)
{
    int h = threadIdx.x;   // 64 threads
    float s = 0.0f;
    for (int k = 0; k < 64; ++k) s += W1[h * 64 + k] * rvec[k];
    pv[h] = s;
}

// pos[n] += (A[1..3][n,:] . pv) * mask — one wave per node, shuffle reduce.
__global__ void __launch_bounds__(256)
vel_update(const float* __restrict__ A, const float* __restrict__ pv,
           const unsigned char* __restrict__ mask, float* __restrict__ pos, int N)
{
    int wid  = (blockIdx.x * blockDim.x + threadIdx.x) >> 5;
    int lane = threadIdx.x & 31;
    if (wid >= N) return;
    const int N64 = N * 64;
    float p0 = pv[lane], p1 = pv[lane + 32];
    const float* a1 = A + N64     + (size_t)wid * 64;
    const float* a2 = A + 2 * N64 + (size_t)wid * 64;
    const float* a3 = A + 3 * N64 + (size_t)wid * 64;
    float s1 = a1[lane] * p0 + a1[lane + 32] * p1;
    float s2 = a2[lane] * p0 + a2[lane + 32] * p1;
    float s3 = a3[lane] * p0 + a3[lane + 32] * p1;
#pragma unroll
    for (int off = 16; off > 0; off >>= 1) {
        s1 += __shfl_down(s1, off);
        s2 += __shfl_down(s2, off);
        s3 += __shfl_down(s3, off);
    }
    if (lane == 0) {
        float mf = mask[wid] ? 1.0f : 0.0f;
        pos[wid * 3 + 0] += s1 * mf;
        pos[wid * 3 + 1] += s2 * mf;
        pos[wid * 3 + 2] += s3 * mf;
    }
}

// One wave per node: labels (5) + dpos (3), shuffle reduction over 64 channels.
__global__ void __launch_bounds__(256)
finalize(const float* __restrict__ h, const float* __restrict__ pos,
         const float* __restrict__ positions, const unsigned char* __restrict__ mask,
         const float* __restrict__ out_w, const float* __restrict__ out_b,
         float* __restrict__ out, int N)
{
    int wid  = (blockIdx.x * blockDim.x + threadIdx.x) >> 5;
    int lane = threadIdx.x & 31;
    if (wid >= N) return;
    float h0 = h[(size_t)wid * 64 + lane];
    float h1 = h[(size_t)wid * 64 + lane + 32];
    float p[5];
#pragma unroll
    for (int k = 0; k < 5; ++k)
        p[k] = h0 * out_w[lane * 6 + k] + h1 * out_w[(lane + 32) * 6 + k];
#pragma unroll
    for (int off = 16; off > 0; off >>= 1) {
#pragma unroll
        for (int k = 0; k < 5; ++k) p[k] += __shfl_down(p[k], off);
    }
    if (lane == 0) {
        float mf = mask[wid] ? 1.0f : 0.0f;
#pragma unroll
        for (int k = 0; k < 5; ++k)
            out[wid * 8 + k] = (p[k] + out_b[k]) * mf;
        out[wid * 8 + 5] = (pos[wid * 3 + 0] - positions[wid * 3 + 0]) * mf;
        out[wid * 8 + 6] = (pos[wid * 3 + 1] - positions[wid * 3 + 1]) * mf;
        out[wid * 8 + 7] = (pos[wid * 3 + 2] - positions[wid * 3 + 2]) * mf;
    }
}

// ---------------------------------------------------------------------------
static void launch_gemm(int flags, int grid, const float* X, const float* W,
                        const float* bias, float* Y, float* Sq, float* Cb,
                        int M, hipStream_t st)
{
    dim3 blk(256);
    switch (flags) {
    case 0:
        gemm64_wmma<0><<<grid, blk, 0, st>>>(X, W, bias, Y, Sq, Cb, M); break;
    case GF_SQCUBE:
        gemm64_wmma<GF_SQCUBE><<<grid, blk, 0, st>>>(X, W, bias, Y, Sq, Cb, M); break;
    case GF_ACC:
        gemm64_wmma<GF_ACC><<<grid, blk, 0, st>>>(X, W, bias, Y, Sq, Cb, M); break;
    case GF_BIAS:
        gemm64_wmma<GF_BIAS><<<grid, blk, 0, st>>>(X, W, bias, Y, Sq, Cb, M); break;
    case GF_BIAS | GF_SILUX:
        gemm64_wmma<GF_BIAS | GF_SILUX><<<grid, blk, 0, st>>>(X, W, bias, Y, Sq, Cb, M); break;
    case GF_ACC | GF_BIAS | GF_SILUX:
        gemm64_wmma<GF_ACC | GF_BIAS | GF_SILUX><<<grid, blk, 0, st>>>(X, W, bias, Y, Sq, Cb, M); break;
    default:
        gemm64_wmma<0><<<grid, blk, 0, st>>>(X, W, bias, Y, Sq, Cb, M); break;
    }
}

extern "C" void kernel_launch(void* const* d_in, const int* in_sizes, int n_in,
                              void* d_out, int out_size, void* d_ws, size_t ws_size,
                              hipStream_t stream)
{
    (void)in_sizes; (void)n_in; (void)out_size; (void)ws_size;
    const int N = 8000, E = 128000;

    const float* positions  = (const float*)d_in[0];
    const float* node_attrs = (const float*)d_in[1];
    const float* t          = (const float*)d_in[2];
    const int*   batch      = (const int*)d_in[3];
    const int*   edge_index = (const int*)d_in[4];
    const float* shifts     = (const float*)d_in[5];
    const unsigned char* mask_env = (const unsigned char*)d_in[6];
    const float* emb_w = (const float*)d_in[7];
    const float* emb_b = (const float*)d_in[8];
    const float* out_w = (const float*)d_in[9];
    const float* out_b = (const float*)d_in[10];
    const float* lin_up = (const float*)d_in[11];
    const float* ew1 = (const float*)d_in[12];
    const float* eb1 = (const float*)d_in[13];
    const float* ew2 = (const float*)d_in[14];
    const float* eb2 = (const float*)d_in[15];
    const float* prod_w = (const float*)d_in[16];
    const float* corr2 = (const float*)d_in[17];
    const float* corr3 = (const float*)d_in[18];
    const float* rw1 = (const float*)d_in[19];
    const float* rb1 = (const float*)d_in[20];
    const float* rw2 = (const float*)d_in[21];
    const float* rb2 = (const float*)d_in[22];
    const float* rvec = (const float*)d_in[23];
    const float* mw1 = (const float*)d_in[24];
    const float* mb1 = (const float*)d_in[25];
    const float* mw2 = (const float*)d_in[26];
    const float* mb2 = (const float*)d_in[27];

    const size_t NH = (size_t)N * 64;
    float* ws    = (float*)d_ws;
    float* h_buf = ws;  ws += NH;
    float* P     = ws;  ws += NH;       // f0
    float* tmp   = ws;  ws += NH;
    float* scal  = ws;  ws += NH;
    float* Ssq   = ws;  ws += NH;
    float* Cu    = ws;  ws += NH;
    float* up    = ws;  ws += NH;
    float* Abuf  = ws;  ws += 4 * NH;   // SH channels 0..3, plane-major
    float* pos   = ws;  ws += (size_t)3 * N;
    float* tnode = ws;  ws += N;
    float* pv    = ws;  ws += 64;

    const dim3 blk(256);
    const int gGemm = (N + 127) / 128;          // 63 blocks

    init_nodes<<<(N * 64 + 255) / 256, blk, 0, stream>>>(
        positions, node_attrs, t, batch, emb_w, emb_b, h_buf, pos, tnode, N);

    const float* node_scal = h_buf;
    for (int i = 0; i < 2; ++i) {
        fill_zero<<<(int)((4 * NH + 255) / 256), blk, 0, stream>>>(Abuf, (int)(4 * NH));

        // up = node_scal @ lin_up[i]
        launch_gemm(0, gGemm, node_scal, lin_up + i * 4096, nullptr, up,
                    nullptr, nullptr, N, stream);

        // fused edge pipeline + scatter (includes 1/AVG_NEIGH)
        edge_scatter<<<2000, blk, 0, stream>>>(edge_index, pos, shifts, tnode,
            ew1 + i * 128, eb1 + i * 64, ew2 + i * 4096, eb2 + i * 64,
            up, Abuf, E, N);

        // f0 = A0 @ prod_w[i,0] (epilogue also emits f0^2, f0^3);
        // then f0 += sq@corr2 + cu@corr3
        launch_gemm(GF_SQCUBE, gGemm, Abuf, prod_w + i * 3 * 4096, nullptr, P,
                    Ssq, Cu, N, stream);
        launch_gemm(GF_ACC, gGemm, Ssq, corr2 + i * 4096, nullptr, P,
                    nullptr, nullptr, N, stream);
        launch_gemm(GF_ACC, gGemm, Cu,  corr3 + i * 4096, nullptr, P,
                    nullptr, nullptr, N, stream);

        // scal = silu(f0@rw1+rb1)@rw2+rb2
        launch_gemm(GF_BIAS, gGemm, P, rw1 + i * 4096, rb1 + i * 64, tmp,
                    nullptr, nullptr, N, stream);
        launch_gemm(GF_BIAS | GF_SILUX, gGemm, tmp, rw2 + i * 4096, rb2 + i * 64,
                    scal, nullptr, nullptr, N, stream);

        // pos += (A[:,:,1:4] . (prod_w[i,1]@rvec)) * mask
        compute_pv<<<1, 64, 0, stream>>>(prod_w + i * 3 * 4096 + 4096, rvec + i * 64, pv);
        vel_update<<<(N * 32 + 255) / 256, blk, 0, stream>>>(Abuf, pv, mask_env, pos, N);

        // h += silu([h,scal]@mw1+mb1)@mw2 + mb2  (mw1 split into two K=64 GEMMs)
        launch_gemm(GF_BIAS, gGemm, h_buf, mw1 + i * 8192, mb1 + i * 64, tmp,
                    nullptr, nullptr, N, stream);
        launch_gemm(GF_ACC,  gGemm, scal,  mw1 + i * 8192 + 4096, nullptr, tmp,
                    nullptr, nullptr, N, stream);
        launch_gemm(GF_ACC | GF_BIAS | GF_SILUX, gGemm, tmp, mw2 + i * 4096,
                    mb2 + i * 64, h_buf, nullptr, nullptr, N, stream);

        node_scal = P;   // node_scal = f0 for next layer
    }

    finalize<<<(N * 32 + 255) / 256, blk, 0, stream>>>(
        h_buf, pos, positions, mask_env, out_w, out_b, (float*)d_out, N);
}
